// GenerativeLanguageModel_42477226557601
// MI455X (gfx1250) — matrix operand order repaired and checked by reading, not compiled, not run on hardware
//
#include <hip/hip_runtime.h>
#include <hip/hip_bf16.h>
#include <math.h>

// ---------------------------------------------------------------------------
// GPT-style LM forward for MI455X (gfx1250, wave32).
//
// Roofline: ~170 GFLOP of GEMM (half in the 1024x768x50304 unembedding) vs a
// 14us HBM floor (330MB weights @ 23.3TB/s). bf16 WMMA (16x16x32, 16384
// FLOP/instr, f32 accumulate) gives ~8x the matrix rate of fp32 WMMA 16x16x4
// and is what the chip rewards. Weights/activations are converted f32->bf16
// on the fly (no extra HBM traffic). 64-row M-tiles (4 WMMA tiles per wave
// per B fragment) give 4x B-reuse; the 154MB embed table is L2-resident
// (192MB L2). All LDS staging uses packed v8bf -> ds_store_b128 (full bank
// width). Sequential log-space scans use wave32 shfl butterflies + HW exp/log.
// Workspace: (2*S*D + S*2304 + 2*S*D) floats = 22,020,096 bytes.
// ---------------------------------------------------------------------------

#define S_LEN 1024
#define D_DIM 768
#define N_LAYERS 12
#define N_HEADS 24
#define DK_DIM 32
#define DV_DIM 32
#define QKV_OUT 2304        // H*(DK+DK+DV)
#define D2_DIM 1536         // 2*D
#define V_SZ 50304
#define MTILES 4            // 4 x 16 = 64 M rows per block
#define AS_STRIDE 776       // 768 + 8 bf16 pad (rows stay 16B aligned)
#define BS_STRIDE 40        // 32 + 8 bf16 pad (16B-aligned fragment reads)

typedef __attribute__((ext_vector_type(8)))  float  v8f;
typedef __attribute__((ext_vector_type(8)))  __bf16 v8bf;
typedef __attribute__((ext_vector_type(16))) __bf16 v16bf;

// ---------------- embedding gather: x[t,:] = embed_w[ids[t],:] --------------
__global__ void embed_gather_k(const int* __restrict__ ids,
                               const float* __restrict__ W,
                               float* __restrict__ x) {
  const int t = blockIdx.x;
  const int tid = threadIdx.x;                 // 256 threads, 3 floats each
  const float* w = W + (size_t)ids[t] * D_DIM;
  float* xr = x + (size_t)t * D_DIM;
  xr[tid]       = w[tid];
  xr[tid + 256] = w[tid + 256];
  xr[tid + 512] = w[tid + 512];
}

// ---------------- bf16 WMMA GEMM, K fixed = 768 -----------------------------
// C[M x N] = A[M x 768] @ B (+bias) (+accumulate). A,B are f32 in memory,
// converted to bf16 on the fly; accumulation in f32 via wmma_f32_16x16x32_bf16.
// Block: 256 thr = 8 waves. Block tile 64(M) x 128(N): each wave owns one
// 16-wide N strip and 4 M tiles (B-fragment reused 4x).
// TRANSB: B is N x 768 row-major (unembedding) -> direct K-contiguous loads.
// else:   B is 768 x N row-major -> 32x128 k-block staged transposed in LDS.
template<bool TRANSB, bool HASBIAS, bool ACCUM>
__global__ __launch_bounds__(256)
void gemm768_bf16(const float* __restrict__ A,
                  const float* __restrict__ B,
                  const float* __restrict__ bias,
                  float* __restrict__ C, int N) {
  __shared__ __bf16 As[64 * AS_STRIDE];
  __shared__ __bf16 Bs[TRANSB ? 8 : 128 * BS_STRIDE];
  const int m0 = blockIdx.y * 64;

  // Stage A: 64 rows x 768 f32 -> bf16 LDS. Each thread owns a contiguous
  // 48-column chunk of a row: packed v8bf -> 6 x ds_store_b128 per row.
  {
    const float* Ab = A + (size_t)m0 * D_DIM;
    const int r0 = threadIdx.x >> 4;            // 0..15
    const int c0 = (threadIdx.x & 15) * 48;     // 16B-aligned chunk base
    for (int rp = 0; rp < 64; rp += 16) {
      const int r = r0 + rp;
      const float* src = Ab + (size_t)r * D_DIM + c0;
      __bf16* dst = &As[r * AS_STRIDE + c0];
#pragma unroll
      for (int i = 0; i < 6; ++i) {
        v8bf pk;
#pragma unroll
        for (int e = 0; e < 8; ++e) pk[e] = (__bf16)src[i * 8 + e];
        *(v8bf*)(dst + i * 8) = pk;
      }
    }
  }
  __syncthreads();

  const int lane = threadIdx.x & 31;
  const int wv   = threadIdx.x >> 5;       // wave id -> 16-wide N strip
  const int l15  = lane & 15;              // M row in tile / N column
  const int hi   = lane >> 4;              // K-half selector (this code's contiguous k = 16*hi + e)
  const int nloc = wv * 16 + l15;
  const int col  = blockIdx.x * 128 + nloc;

  const v8f vzero = {0.f, 0.f, 0.f, 0.f, 0.f, 0.f, 0.f, 0.f};
  v8f acc[MTILES];
#pragma unroll
  for (int i = 0; i < MTILES; ++i) acc[i] = vzero;

  for (int k0 = 0; k0 < D_DIM; k0 += 32) {
    // ---- B fragment: 32(K) x 16(N); lane element e -> K = hi*16+e, N = l15.
    v16bf bf;
    if (TRANSB) {
      // B row-major N x 768: 16 consecutive f32 per lane (64B, L2-friendly).
      const float* bp = B + (size_t)col * D_DIM + k0 + hi * 16;
#pragma unroll
      for (int e = 0; e < 16; ++e) bf[e] = (__bf16)bp[e];
    } else {
      __syncthreads();   // previous iteration done reading Bs
      {
        // Stage B[k0..k0+31][colBase..+127] transposed to Bs[n][k] (bf16).
        // Thread (n, k-half) converts 16 k-values of column n; global reads
        // stay coalesced across the 128 n-threads; 2 packed b128 LDS stores.
        const int n  = threadIdx.x & 127;
        const int k2 = threadIdx.x >> 7;   // 0/1
        const float* Bb = B + (size_t)(k0 + k2 * 16) * N + blockIdx.x * 128 + n;
        __bf16* dst = &Bs[n * BS_STRIDE + k2 * 16];
        v8bf p0, p1;
#pragma unroll
        for (int e = 0; e < 8; ++e) p0[e] = (__bf16)Bb[(size_t)e * N];
#pragma unroll
        for (int e = 0; e < 8; ++e) p1[e] = (__bf16)Bb[(size_t)(8 + e) * N];
        *(v8bf*)dst = p0;
        *(v8bf*)(dst + 8) = p1;
      }
      __syncthreads();
      const v8bf blo = *(const v8bf*)&Bs[nloc * BS_STRIDE + hi * 16];
      const v8bf bhi = *(const v8bf*)&Bs[nloc * BS_STRIDE + hi * 16 + 8];
      bf = __builtin_shufflevector(blo, bhi,
                                   0, 1, 2, 3, 4, 5, 6, 7,
                                   8, 9, 10, 11, 12, 13, 14, 15);
    }
    // ---- A fragments (16-bit 16x32 layout): lane m = l15; element e ->
    // K = k0+hi*16+e (matches B fragment order). Two aligned
    // 16B LDS reads per tile. One WMMA per M tile, B fragment reused 4x.
#pragma unroll
    for (int mt = 0; mt < MTILES; ++mt) {
      const int m = mt * 16 + l15;
      const v8bf alo = *(const v8bf*)&As[m * AS_STRIDE + k0 + hi * 16];
      const v8bf ahi = *(const v8bf*)&As[m * AS_STRIDE + k0 + hi * 16 + 8];
      const v16bf af = __builtin_shufflevector(alo, ahi,
                                               0, 1, 2, 3, 4, 5, 6, 7,
                                               8, 9, 10, 11, 12, 13, 14, 15);
      acc[mt] = __builtin_amdgcn_wmma_f32_16x16x32_bf16(
          false, af, false, bf, (short)0, acc[mt], false, false);
    }
  }

  float bv = 0.f;
  if (HASBIAS) bv = bias[col];
  // C/D layout: VGPR r -> row (tile base) + r + 8*hi, col = l15.
#pragma unroll
  for (int mt = 0; mt < MTILES; ++mt) {
    const int rbase = m0 + mt * 16 + hi * 8;
#pragma unroll
    for (int r = 0; r < 8; ++r) {
      const size_t off = (size_t)(rbase + r) * N + col;
      const float v = acc[mt][r] + bv;
      if (ACCUM) C[off] += v; else C[off] = v;
    }
  }
}

// ---------------- position recurrence (sequential over S per channel) -------
// h = x@pos_w+pos_b precomputed (S x 1536). Running logsumexp per channel.
__global__ void posrec_k(const float* __restrict__ h, float* __restrict__ x) {
  const int d = blockIdx.x * blockDim.x + threadIdx.x;
  if (d >= D_DIM) return;
  float astar = 0.f;
  float m = 0.f, s = 1.f;   // state after prepended j=0 term (value 0)
  for (int t = 0; t < S_LEN; ++t) {
    const float* ht = h + (size_t)t * D2_DIM;
    const float lc = ht[d];            // pre-logsigmoid coeff
    const float lg = ht[D_DIM + d];    // logit
    const float ls = (lc >= 0.f) ? -log1pf(__expf(-lc))
                                 : (lc - log1pf(__expf(lc)));
    astar += ls;                       // a_star[t+1]
    const float v = lg - astar;        // p[t+1] - a_star[t+1]
    const float nm = fmaxf(m, v);
    s = s * __expf(m - nm) + __expf(v - nm);
    m = nm;
    x[(size_t)t * D_DIM + d] += astar + m + __logf(s);  // += rec[t+1]
  }
}

// ---------------- layernorm: one block (256 thr) per row --------------------
__global__ __launch_bounds__(256)
void layernorm_k(const float* __restrict__ x, const float* __restrict__ g,
                 const float* __restrict__ b, float* __restrict__ out) {
  __shared__ float red[8];
  __shared__ float stat[2];
  const int row = blockIdx.x;
  const int tid = threadIdx.x;
  const float* xr = x + (size_t)row * D_DIM;
  const float v0 = xr[tid], v1 = xr[tid + 256], v2 = xr[tid + 512];
  float s = v0 + v1 + v2;
  for (int o = 16; o > 0; o >>= 1) s += __shfl_xor(s, o, 32);
  if ((tid & 31) == 0) red[tid >> 5] = s;
  __syncthreads();
  if (tid == 0) {
    float tot = 0.f;
    for (int i = 0; i < 8; ++i) tot += red[i];
    stat[0] = tot * (1.0f / D_DIM);
  }
  __syncthreads();
  const float mean = stat[0];
  const float d0 = v0 - mean, d1 = v1 - mean, d2 = v2 - mean;
  float q = d0 * d0 + d1 * d1 + d2 * d2;
  for (int o = 16; o > 0; o >>= 1) q += __shfl_xor(q, o, 32);
  if ((tid & 31) == 0) red[tid >> 5] = q;
  __syncthreads();
  if (tid == 0) {
    float tot = 0.f;
    for (int i = 0; i < 8; ++i) tot += red[i];
    stat[1] = rsqrtf(tot * (1.0f / D_DIM) + 1e-5f);
  }
  __syncthreads();
  const float rs = stat[1];
  float* orow = out + (size_t)row * D_DIM;
  orow[tid]       = d0 * rs * g[tid]       + b[tid];
  orow[tid + 256] = d1 * rs * g[tid + 256] + b[tid + 256];
  orow[tid + 512] = d2 * rs * g[tid + 512] + b[tid + 512];
}

// ---------------- log-space linear attention scan ---------------------------
// One block per head; 1024 threads = 32 waves. Wave j owns DV column j, lane
// i owns DK index i. H_S state lives in registers; logsumexp over DK is a
// pure intra-wave shfl butterfly (wave32). H_Z state replicated per wave.
__global__ __launch_bounds__(1024)
void attn_scan_k(const float* __restrict__ y, float* __restrict__ att) {
  const int head = blockIdx.x;
  const int lane = threadIdx.x & 31;   // i : DK index
  const int j    = threadIdx.x >> 5;   // DV index (one wave per j)
  const float* yh = y + head * (DK_DIM + DK_DIM + DV_DIM);
  float Ms = -1e30f, Ss = 0.f;         // running lse state for H_S[i][j]
  float Mz = -1e30f, Sz = 0.f;         // running lse state for H_Z[i]
  for (int t = 0; t < S_LEN; ++t) {
    const float* yt = yh + (size_t)t * QKV_OUT;
    const float q = yt[lane];
    const float k = yt[DK_DIM + lane];
    const float v = yt[2 * DK_DIM + j];
    const float a = k + v;
    float nm = fmaxf(Ms, a);
    Ss = Ss * __expf(Ms - nm) + __expf(a - nm);
    Ms = nm;
    nm = fmaxf(Mz, k);
    Sz = Sz * __expf(Mz - nm) + __expf(k - nm);
    Mz = nm;
    const float hs = q + Ms + __logf(Ss);
    const float hz = q + Mz + __logf(Sz);
    float m1 = hs, m2 = hz;
    for (int o = 16; o > 0; o >>= 1) {
      m1 = fmaxf(m1, __shfl_xor(m1, o, 32));
      m2 = fmaxf(m2, __shfl_xor(m2, o, 32));
    }
    float e1 = __expf(hs - m1);
    float e2 = __expf(hz - m2);
    for (int o = 16; o > 0; o >>= 1) {
      e1 += __shfl_xor(e1, o, 32);
      e2 += __shfl_xor(e2, o, 32);
    }
    if (lane == 0)
      att[(size_t)t * D_DIM + head * DV_DIM + j] =
          (m1 + __logf(e1)) - (m2 + __logf(e2));
  }
}

// ---------------- GLU: g = z[:, :768] * sigmoid(z[:, 768:]) -----------------
__global__ void glu_k(const float* __restrict__ z, float* __restrict__ g) {
  const int idx = blockIdx.x * blockDim.x + threadIdx.x;
  const int t = idx / D_DIM;
  const int d = idx - t * D_DIM;
  const float a  = z[(size_t)t * D2_DIM + d];
  const float bb = z[(size_t)t * D2_DIM + D_DIM + d];
  g[idx] = a / (1.f + __expf(-bb));
}

// ---------------------------------------------------------------------------
extern "C" void kernel_launch(void* const* d_in, const int* in_sizes, int n_in,
                              void* d_out, int out_size, void* d_ws, size_t ws_size,
                              hipStream_t stream) {
  (void)in_sizes; (void)n_in; (void)out_size; (void)ws_size;
  const int*   tok  = (const int*)  d_in[0];
  const float* embw = (const float*)d_in[1];
  const float* posw = (const float*)d_in[2];
  const float* posb = (const float*)d_in[3];
  const float* ln1g = (const float*)d_in[4];
  const float* ln1b = (const float*)d_in[5];
  const float* qkvw = (const float*)d_in[6];
  const float* qkvb = (const float*)d_in[7];
  const float* ffw1 = (const float*)d_in[8];
  const float* ffb1 = (const float*)d_in[9];
  const float* ffw2 = (const float*)d_in[10];
  const float* lnfg = (const float*)d_in[11];
  const float* lnfb = (const float*)d_in[12];
  float* out = (float*)d_out;
  float* ws  = (float*)d_ws;

  float* x   = ws;                              // S*D  residual stream
  float* xn  = x   + (size_t)S_LEN * D_DIM;     // S*D  layernorm output
  float* y   = xn  + (size_t)S_LEN * D_DIM;     // S*2304 (pos h / qkv / ff1 z)
  float* att = y   + (size_t)S_LEN * QKV_OUT;   // S*D
  float* g   = att + (size_t)S_LEN * D_DIM;     // S*D

  // x = embed_w[token_ids]
  embed_gather_k<<<S_LEN, 256, 0, stream>>>(tok, embw, x);

  // h = x @ pos_w + pos_b ; x += positional recurrence
  gemm768_bf16<false, true, false><<<dim3(D2_DIM / 128, S_LEN / 64), 256, 0, stream>>>(
      x, posw, posb, y, D2_DIM);
  posrec_k<<<3, 256, 0, stream>>>(y, x);

  for (int l = 0; l < N_LAYERS; ++l) {
    layernorm_k<<<S_LEN, 256, 0, stream>>>(
        x, ln1g + (size_t)l * D_DIM, ln1b + (size_t)l * D_DIM, xn);
    gemm768_bf16<false, true, false><<<dim3(QKV_OUT / 128, S_LEN / 64), 256, 0, stream>>>(
        xn, qkvw + (size_t)l * D_DIM * QKV_OUT, qkvb + (size_t)l * QKV_OUT, y, QKV_OUT);
    attn_scan_k<<<N_HEADS, 1024, 0, stream>>>(y, att);
    gemm768_bf16<false, true, false><<<dim3(D2_DIM / 128, S_LEN / 64), 256, 0, stream>>>(
        att, ffw1 + (size_t)l * D_DIM * D2_DIM, ffb1 + (size_t)l * D2_DIM, y, D2_DIM);
    glu_k<<<(S_LEN * D_DIM) / 256, 256, 0, stream>>>(y, g);
    // x += glu @ ff_w2  (residual fused via ACCUM epilogue)
    gemm768_bf16<false, false, true><<<dim3(D_DIM / 128, S_LEN / 64), 256, 0, stream>>>(
        g, ffw2 + (size_t)l * D_DIM * D_DIM, nullptr, x, D_DIM);
  }

  // final layernorm + unembedding (B = embed_w is N x K row-major: TRANSB)
  layernorm_k<<<S_LEN, 256, 0, stream>>>(x, lnfg, lnfb, xn);
  gemm768_bf16<true, false, false><<<dim3(V_SZ / 128, S_LEN / 64), 256, 0, stream>>>(
      xn, embw, nullptr, out, V_SZ);
}
